// FP8DynamicLinear_31559419691839
// MI455X (gfx1250) — 8-bit matrix operand order repaired and checked by reading, compiled, not run on hardware
//
#include <hip/hip_runtime.h>
#include <hip/hip_bf16.h>

typedef __attribute__((ext_vector_type(16))) int   v16i;
typedef __attribute__((ext_vector_type(8)))  float v8f;

#define FP8_MAX_F 448.0f


__device__ __forceinline__ float bf16_rne(float f) {
  unsigned u = __float_as_uint(f);
  u += 0x7fffu + ((u >> 16) & 1u);
  return __uint_as_float(u & 0xffff0000u);
}

__device__ __forceinline__ float bf16_abs(float f) {
  unsigned u = __float_as_uint(f);
  u += 0x7fffu + ((u >> 16) & 1u);
  return __uint_as_float(u & 0x7fff0000u);
}

__device__ __forceinline__ unsigned char f32_to_e4m3(float f) {
  float a = fminf(fabsf(f), FP8_MAX_F);
  unsigned sign = (__float_as_uint(f) >> 24) & 0x80u;
  if (a < 0.015625f) {
    int q = (int)rintf(a * 512.0f);
    return (unsigned char)(sign | (unsigned)q);
  }
  unsigned u = __float_as_uint(a);
  u += 0x7ffffu + ((u >> 20) & 1u);
  unsigned exp8 = (u >> 23) & 0xffu;
  unsigned man3 = (u >> 20) & 7u;
  unsigned code = ((exp8 - 120u) << 3) | man3;
  return (unsigned char)(sign | code);
}

__device__ __forceinline__ void async_b128(const void* gptr, const void* lptr) {
  unsigned lds = (unsigned)(unsigned long long)lptr;
  unsigned long long g = (unsigned long long)gptr;
  asm volatile("global_load_async_to_lds_b128 %0, %1, off" :: "v"(lds), "v"(g) : "memory");
}
__device__ __forceinline__ void wait_async0() {
  asm volatile("s_wait_asynccnt 0" ::: "memory");
}

__global__ void init_amax_kernel(unsigned* amax_bits) {
  if (threadIdx.x == 0) *amax_bits = 0u;
}

__global__ __launch_bounds__(256) void amax_kernel(const float* __restrict__ x,
                                                   long n4, unsigned* __restrict__ amax_bits) {
  const float4* xv = (const float4*)x;
  float m = 0.0f;
  long stride = (long)gridDim.x * blockDim.x;
  for (long i = (long)blockIdx.x * blockDim.x + threadIdx.x; i < n4; i += stride) {
    float4 v = xv[i];
    m = fmaxf(m, bf16_abs(v.x));
    m = fmaxf(m, bf16_abs(v.y));
    m = fmaxf(m, bf16_abs(v.z));
    m = fmaxf(m, bf16_abs(v.w));
  }
  __shared__ float red[256];
  int tid = threadIdx.x;
  red[tid] = m;
  __syncthreads();
  for (int s = 128; s > 0; s >>= 1) {
    if (tid < s) red[tid] = fmaxf(red[tid], red[tid + s]);
    __syncthreads();
  }
  if (tid == 0) atomicMax(amax_bits, __float_as_uint(red[0]));
}

__global__ __launch_bounds__(256) void quant_w_kernel(const float* __restrict__ w,
                                                      unsigned char* __restrict__ qw, long n16) {
  long i = (long)blockIdx.x * blockDim.x + threadIdx.x;
  if (i >= n16) return;
  const float4* wv = (const float4*)w;
  unsigned char b[16];
#pragma unroll
  for (int j = 0; j < 4; ++j) {
    float4 v = wv[i * 4 + j];
    b[j * 4 + 0] = f32_to_e4m3(v.x);
    b[j * 4 + 1] = f32_to_e4m3(v.y);
    b[j * 4 + 2] = f32_to_e4m3(v.z);
    b[j * 4 + 3] = f32_to_e4m3(v.w);
  }
  ((uint4*)qw)[i] = *(const uint4*)b;
}

__global__ __launch_bounds__(256) void quant_x_kernel(const float* __restrict__ x,
                                                      unsigned char* __restrict__ qx,
                                                      const float* __restrict__ amax,
                                                      long n16) {
  long i = (long)blockIdx.x * blockDim.x + threadIdx.x;
  if (i >= n16) return;
  float xs = fmaxf(amax[0], 1e-12f) * (1.0f / FP8_MAX_F);
  float inv = 1.0f / xs;
  const float4* xv = (const float4*)x;
  unsigned char b[16];
#pragma unroll
  for (int j = 0; j < 4; ++j) {
    float4 v = xv[i * 4 + j];
    b[j * 4 + 0] = f32_to_e4m3(bf16_rne(v.x) * inv);
    b[j * 4 + 1] = f32_to_e4m3(bf16_rne(v.y) * inv);
    b[j * 4 + 2] = f32_to_e4m3(bf16_rne(v.z) * inv);
    b[j * 4 + 3] = f32_to_e4m3(bf16_rne(v.w) * inv);
  }
  ((uint4*)qx)[i] = *(const uint4*)b;
}

__global__ __launch_bounds__(256) void gemm_fp8_kernel(
    const unsigned char* __restrict__ qa, const unsigned char* __restrict__ qb,
    const float* __restrict__ amax, const float* __restrict__ wscale,
    const float* __restrict__ bias, __hip_bfloat16* __restrict__ out,
    int M, int N, int K) {
  constexpr int TM = 128, TN = 64, TK = 128, LDA = 144;
  __shared__ alignas(16) unsigned char sA[2][TM * LDA];
  __shared__ alignas(16) unsigned char sB[2][TN * LDA];

  const int tid  = threadIdx.x;
  const int wave = tid >> 5;
  const int lane = tid & 31;
  const int hi   = lane >> 4;
  const int lm   = lane & 15;

  const long bm = (long)blockIdx.y * TM;
  const long bn = (long)blockIdx.x * TN;
  const unsigned char* Ag = qa + bm * K;
  const unsigned char* Bg = qb + bn * K;
  const int NK = K / TK;

  auto issue = [&](int buf, int kt) {
    const long kbase = (long)kt * TK;
#pragma unroll
    for (int i = 0; i < 4; ++i) {
      int c = tid * 4 + i;
      int row = c >> 3, cc = c & 7;
      async_b128(Ag + (long)row * K + kbase + cc * 16, &sA[buf][row * LDA + cc * 16]);
    }
#pragma unroll
    for (int i = 0; i < 2; ++i) {
      int c = tid * 2 + i;
      int row = c >> 3, cc = c & 7;
      async_b128(Bg + (long)row * K + kbase + cc * 16, &sB[buf][row * LDA + cc * 16]);
    }
  };

  v8f acc[4] = {};

  issue(0, 0);
  for (int kt = 0; kt < NK; ++kt) {
    const int cur = kt & 1;
    wait_async0();
    __syncthreads();
    if (kt + 1 < NK) issue(cur ^ 1, kt + 1);

    const unsigned char* pa = &sA[cur][(wave * 16 + lm) * LDA + hi * 16];
    v16i af;
#pragma unroll
    for (int h = 0; h < 2; ++h)
#pragma unroll
      for (int c = 0; c < 4; ++c) {
        unsigned long long q = *(const unsigned long long*)(pa + h * 64 + (c >> 1) * 32 + (c & 1) * 8);
        af[h * 8 + c * 2]     = (int)(unsigned)(q & 0xffffffffu);
        af[h * 8 + c * 2 + 1] = (int)(unsigned)(q >> 32);
      }

#pragma unroll
    for (int t = 0; t < 4; ++t) {
      const unsigned char* pb = &sB[cur][(t * 16 + lm) * LDA + hi * 16];
      v16i bfrag;
#pragma unroll
      for (int g = 0; g < 4; ++g) {
        int4 q = *(const int4*)(pb + g * 32);
        bfrag[4 * g + 0] = q.x;
        bfrag[4 * g + 1] = q.y;
        bfrag[4 * g + 2] = q.z;
        bfrag[4 * g + 3] = q.w;
      }
      acc[t] = __builtin_amdgcn_wmma_f32_16x16x128_fp8_fp8(af, bfrag, (short)0, acc[t],
                                                           false, false);
    }
    __syncthreads();
  }

  const float xs = fmaxf(amax[0], 1e-12f) * (1.0f / FP8_MAX_F);
  const float s  = xs * wscale[0];
#pragma unroll
  for (int t = 0; t < 4; ++t) {
    const int col = (int)bn + t * 16 + lm;
    const float bc = bias[col];
#pragma unroll
    for (int r = 0; r < 8; ++r) {
      const int row = (int)bm + wave * 16 + r + hi * 8;
      out[(long)row * N + col] = __float2bfloat16(acc[t][r] * s + bc);
    }
  }
}

extern "C" void kernel_launch(void* const* d_in, const int* in_sizes, int n_in,
                              void* d_out, int out_size, void* d_ws, size_t ws_size,
                              hipStream_t stream) {
  const float* x      = (const float*)d_in[0];
  const float* weight = (const float*)d_in[1];
  const float* wscale = (const float*)d_in[2];
  const float* bias   = (const float*)d_in[3];
  __hip_bfloat16* out = (__hip_bfloat16*)d_out;

  const long xN = (long)in_sizes[0];
  const long wN = (long)in_sizes[1];
  const int  N  = in_sizes[3];
  const int  K  = (int)(wN / N);
  const int  M  = (int)(xN / K);

  unsigned char* qx = (unsigned char*)d_ws;
  unsigned char* qw = qx + (size_t)M * (size_t)K;
  float* amax = (float*)(qw + (size_t)N * (size_t)K);

  init_amax_kernel<<<1, 32, 0, stream>>>((unsigned*)amax);
  amax_kernel<<<2048, 256, 0, stream>>>(x, xN / 4, (unsigned*)amax);

  const long w16 = wN / 16;
  quant_w_kernel<<<(unsigned)((w16 + 255) / 256), 256, 0, stream>>>(weight, qw, w16);
  const long x16 = xN / 16;
  quant_x_kernel<<<(unsigned)((x16 + 255) / 256), 256, 0, stream>>>(x, qx, amax, x16);

  dim3 grid((unsigned)(N / 64), (unsigned)(M / 128));
  gemm_fp8_kernel<<<grid, 256, 0, stream>>>(qx, qw, amax, wscale, bias, out, M, N, K);
}
